// IRAgent_76484777607571
// MI455X (gfx1250) — matrix operand order repaired and checked by reading, not compiled, not run on hardware
//
#include <hip/hip_runtime.h>
#include <hip/hip_bf16.h>

// ---------------------------------------------------------------------------
// Problem constants (from reference)
// ---------------------------------------------------------------------------
#define BB   1024
#define NA   10
#define NE   10
#define NAL  9
#define BNA  (BB * NA)            // 10240
#define ME   (BNA * NE * 2)       // 204800  (enemy stream rows)
#define MA   (BNA * NAL * 2)      // 184320  (ally stream rows)
#define MV   (BNA * 2)            // 20480   (vf rows)
#define MAT  (BNA * NE)           // 102400  (attention rows)
#define QEPS 1e-5f

typedef __attribute__((ext_vector_type(16))) _Float16 v16h;
typedef __attribute__((ext_vector_type(8)))  _Float16 v8h;
typedef __attribute__((ext_vector_type(8)))  float    v8f;

// ---------------------------------------------------------------------------
// Generic WMMA GEMM: Y[M x N] = X[M x K(f16)] @ W (packed transposed [N][K] f16)
//   - block = 128 threads (4 wave32), each wave -> one 16x16 D tile
//   - grid  = (M/16, ceil(N/64))
//   - optional fused bias + relu, optional f32 / f16 output,
//     optional per-column sum-of-squares (qbn statistic) via atomics.
// ---------------------------------------------------------------------------
__global__ __launch_bounds__(128) void wmma_gemm(
    const _Float16* __restrict__ X, const _Float16* __restrict__ Wt,
    float* __restrict__ Y, _Float16* __restrict__ Yh,
    const float* __restrict__ bias, float* __restrict__ colsq,
    int M, int K, int N, int doRelu)
{
  const int lane = threadIdx.x & 31;
  const int wave = threadIdx.x >> 5;
  const int rb   = blockIdx.x * 16;
  const int ntile = blockIdx.y * 4 + wave;
  if (ntile * 16 >= N) return;
  const int nb = ntile * 16;
  const int l  = lane & 15;
  const int hi = lane >> 4;

  v8f c = {};
  // A fragment source: row rb+l, element halves split by hi (K 0-7/16-23 vs 8-15/24-31)
  const _Float16* xr = X  + (size_t)(rb + l) * K + hi * 8;
  // B fragment source: column nb+l of W == row of Wt, element halves split by hi (K 0-7/16-23 vs 8-15/24-31)
  const _Float16* wr = Wt + (size_t)(nb + l) * K + hi * 8;

  for (int kb = 0; kb < K; kb += 32) {
    __builtin_prefetch(xr + kb + 32, 0, 1);   // global_prefetch on A stream
    union { v16h v; v8h h[2]; } a, b;
    a.h[0] = *(const v8h*)(xr + kb);          // K = kb+hi*8   .. +7
    a.h[1] = *(const v8h*)(xr + kb + 16);     // K = kb+16+hi*8.. +7
    b.h[0] = *(const v8h*)(wr + kb);          // K = kb+hi*8   .. +7
    b.h[1] = *(const v8h*)(wr + kb + 16);     // K = kb+16+hi*8.. +7
    c = __builtin_amdgcn_wmma_f32_16x16x32_f16(
            false, a.v, false, b.v, (short)0, c, false, false);
  }

  const float bcol = bias ? bias[nb + l] : 0.0f;
  float partial = 0.0f;
  for (int r = 0; r < 8; ++r) {
    const int row = rb + hi * 8 + r;           // C/D layout: lanes>=16 hold M=8..15
    float v = c[r] + bcol;
    if (doRelu) v = fmaxf(v, 0.0f);
    partial += v * v;
    const size_t off = (size_t)row * N + nb + l;
    if (Y)  Y[off]  = v;
    if (Yh) Yh[off] = (_Float16)v;
  }
  if (colsq) atomicAdd(&colsq[nb + l], partial);
}

// ---------------------------------------------------------------------------
// qbn (global per-channel RMS) + qrelu (pairwise modulus clamp) fused epilogue.
// Rows come in adjacent pairs (the size-2 axis). colsq[c] holds sum(y^2) over M rows.
// ---------------------------------------------------------------------------
__global__ void qbn_qrelu(const float* __restrict__ Y, const float* __restrict__ colsq,
                          int M, int N, int Nvalid,
                          _Float16* __restrict__ outH, float* __restrict__ out32)
{
  const int tid = blockIdx.x * blockDim.x + threadIdx.x;
  const int total = (M / 2) * Nvalid;
  if (tid >= total) return;
  const int j = tid / Nvalid;
  const int c = tid % Nvalid;
  const float inv = rsqrtf(colsq[c] / (float)M + QEPS);
  const size_t r0 = (size_t)(2 * j) * N + c;
  const size_t r1 = r0 + N;
  float y0 = Y[r0] * inv;
  float y1 = Y[r1] * inv;
  const float mod   = sqrtf(y0 * y0 + y1 * y1);
  const float coeff = mod / fmaxf(1.0f, mod);
  y0 *= coeff; y1 *= coeff;
  if (outH)  { outH[r0] = (_Float16)y0; outH[r1] = (_Float16)y1; }
  if (out32) { out32[(size_t)(2 * j) * Nvalid + c] = y0;
               out32[(size_t)(2 * j + 1) * Nvalid + c] = y1; }
}

// ---------------------------------------------------------------------------
// build_vec: feats (.., 9) + own (.., 11) -> two rows of 19 features, K-padded to 32
// ---------------------------------------------------------------------------
__global__ void build_vec_k(const float* __restrict__ feats, const float* __restrict__ own,
                            _Float16* __restrict__ out, int nEnt, int total)
{
  const int t = blockIdx.x * blockDim.x + threadIdx.x;
  if (t >= total) return;
  const int bna = t / nEnt;
  const float* f = feats + (size_t)t * 9;
  const float* o = own + (size_t)bna * 11;
  const float p0 = f[2], p1 = f[3];
  const float pm = sqrtf(p0 * p0 + p1 * p1);
  const bool  mask  = (pm == 0.0f);
  const float denom = mask ? 1.0f : pm;
  _Float16* o0 = out + (size_t)(2 * t) * 32;
  _Float16* o1 = o0 + 32;
  o0[0] = (_Float16)p0; o1[0] = (_Float16)p1;
  const float other[7] = { f[0], f[1], f[4], f[5], f[6], f[7], f[8] };
  for (int j = 0; j < 7; ++j) {
    const float cf = mask ? 0.0f : sqrtf(2.0f * other[j] * other[j]) / denom;
    o0[1 + j] = (_Float16)(p0 * cf);
    o1[1 + j] = (_Float16)(p1 * cf);
  }
  for (int k = 0; k < 11; ++k) {
    const float oc = mask ? 0.0f : sqrtf(2.0f * o[k] * o[k]) / denom;
    o0[8 + k] = (_Float16)(p0 * oc);
    o1[8 + k] = (_Float16)(p1 * oc);
  }
  for (int c2 = 19; c2 < 32; ++c2) { o0[c2] = (_Float16)0.0f; o1[c2] = (_Float16)0.0f; }
}

// qmaxpool over 19 entities (enemy first, then ally) per (b,na,channel)
__global__ void qmaxpool_k(const _Float16* __restrict__ e_rc, const _Float16* __restrict__ a_rc,
                           _Float16* __restrict__ pooled, int total)
{
  const int t = blockIdx.x * blockDim.x + threadIdx.x;
  if (t >= total) return;
  const int c = t & 63;
  const int bna = t >> 6;
  float best = -1.0f, bv0 = 0.0f, bv1 = 0.0f;
  for (int ne = 0; ne < NE; ++ne) {
    const size_t b = ((size_t)(bna * NE + ne) * 2) * 64 + c;
    const float x0 = (float)e_rc[b], x1 = (float)e_rc[b + 64];
    const float m = x0 * x0 + x1 * x1;
    if (m > best) { best = m; bv0 = x0; bv1 = x1; }
  }
  for (int na2 = 0; na2 < NAL; ++na2) {
    const size_t b = ((size_t)(bna * NAL + na2) * 2) * 64 + c;
    const float x0 = (float)a_rc[b], x1 = (float)a_rc[b + 64];
    const float m = x0 * x0 + x1 * x1;
    if (m > best) { best = m; bv0 = x0; bv1 = x1; }
  }
  const size_t ob = ((size_t)bna * 2) * 64 + c;
  pooled[ob] = (_Float16)bv0; pooled[ob + 64] = (_Float16)bv1;
}

// sf = qmerge(vf): per (bna, c) sum over the pair
__global__ void qmerge_sf_k(const _Float16* __restrict__ vf, float* __restrict__ sf,
                            _Float16* __restrict__ sfh, int total)
{
  const int t = blockIdx.x * blockDim.x + threadIdx.x;
  if (t >= total) return;
  const int c = t & 63;
  const int bna = t >> 6;
  const size_t b = ((size_t)bna * 2) * 64 + c;
  const float v0 = (float)vf[b], v1 = (float)vf[b + 64];
  const float s = v0 * v0 + v1 * v1;
  sf[(size_t)bna * 64 + c] = s;
  sfh[(size_t)bna * 64 + c] = (_Float16)s;
}

// cat = [vf broadcast | e_rc]  (Me x 128)
__global__ void build_cat_k(const _Float16* __restrict__ vf, const _Float16* __restrict__ e_rc,
                            _Float16* __restrict__ cat, long total)
{
  const long t = (long)blockIdx.x * blockDim.x + threadIdx.x;
  if (t >= total) return;
  const int  col = (int)(t & 127);
  const long row = t >> 7;                    // (bna*NE + ne)*2 + p
  const int  p   = (int)(row & 1);
  const long ent = row >> 1;
  const long bna = ent / NE;
  _Float16 v;
  if (col < 64) v = vf[((size_t)(bna * 2 + p)) * 64 + col];
  else          v = e_rc[(size_t)row * 64 + (col - 64)];
  cat[t] = v;
}

// atf = [sf broadcast | e_scalar]  (Mat x 128)
__global__ void build_atf_k(const _Float16* __restrict__ sfh, const _Float16* __restrict__ e_rc,
                            _Float16* __restrict__ atf, long total)
{
  const long t = (long)blockIdx.x * blockDim.x + threadIdx.x;
  if (t >= total) return;
  const int  col = (int)(t & 127);
  const long row = t >> 7;                    // bna*NE + ne
  if (col < 64) {
    atf[t] = sfh[(size_t)(row / NE) * 64 + col];
  } else {
    const size_t eb = (size_t)row * 2 * 64 + (col - 64);
    const float e0 = (float)e_rc[eb], e1 = (float)e_rc[eb + 64];
    atf[t] = (_Float16)(e0 * e0 + e1 * e1);
  }
}

// move_vec[bna][p] = sum over entities of (e_imp + a_imp)
__global__ void movevec_k(const float* __restrict__ e_imp, const float* __restrict__ a_imp,
                          float* __restrict__ mv, int total)
{
  const int t = blockIdx.x * blockDim.x + threadIdx.x;
  if (t >= total) return;
  const int p = t & 1;
  const int bna = t >> 1;
  float s = 0.0f;
  for (int ne = 0; ne < NE; ++ne) {
    const int r = (bna * NE + ne) * 2 + p;
    s += e_imp[r] + a_imp[r];
  }
  mv[t] = s;
}

// final assemble: [oaq0, oaq1, mm+mv1, mm-mv1, mm+mv0, mm-mv0, aq(10)]
__global__ void final_k(const float* __restrict__ oaq, const float* __restrict__ mv,
                        const float* __restrict__ aq, float* __restrict__ out, int total)
{
  const int t = blockIdx.x * blockDim.x + threadIdx.x;
  if (t >= total) return;
  const float o0 = oaq[(size_t)t * 16 + 0];
  const float o1 = oaq[(size_t)t * 16 + 1];
  const float mm = oaq[(size_t)t * 16 + 2];
  const float m0 = mv[t * 2 + 0], m1 = mv[t * 2 + 1];
  float* ob = out + (size_t)t * 16;
  ob[0] = o0; ob[1] = o1;
  ob[2] = mm + m1; ob[3] = mm - m1;
  ob[4] = mm + m0; ob[5] = mm - m0;
  for (int ne = 0; ne < NE; ++ne)
    ob[6 + ne] = aq[((size_t)t * NE + ne) * 16 + 0];
}

// weight repack: f32 [K][N] -> f16 transposed zero-padded [Np][Kp]
__global__ void pack_w_k(const float* __restrict__ W, _Float16* __restrict__ Wt,
                         int K, int N, int Kp, int Np)
{
  const int t = blockIdx.x * blockDim.x + threadIdx.x;
  if (t >= Kp * Np) return;
  const int k = t % Kp;
  const int n = t / Kp;
  const _Float16 v = (k < K && n < N) ? (_Float16)W[(size_t)k * N + n] : (_Float16)0.0f;
  Wt[(size_t)n * Kp + k] = v;
}

__global__ void pack_b_k(const float* __restrict__ b, float* __restrict__ bp, int n, int np)
{
  const int t = blockIdx.x * blockDim.x + threadIdx.x;
  if (t >= np) return;
  bp[t] = (t < n) ? b[t] : 0.0f;
}

__global__ void zero_f32_k(float* __restrict__ p, int n)
{
  const int t = blockIdx.x * blockDim.x + threadIdx.x;
  if (t < n) p[t] = 0.0f;
}

// ---------------------------------------------------------------------------
// Host orchestration
// ---------------------------------------------------------------------------
static inline int cdiv(long a, int b) { return (int)((a + b - 1) / b); }

extern "C" void kernel_launch(void* const* d_in, const int* in_sizes, int n_in,
                              void* d_out, int out_size, void* d_ws, size_t ws_size,
                              hipStream_t stream)
{
  (void)in_sizes; (void)n_in; (void)out_size; (void)ws_size;
  const float* own   = (const float*)d_in[0];
  const float* enemy = (const float*)d_in[1];
  const float* ally  = (const float*)d_in[2];
  const float* W_e1  = (const float*)d_in[3];
  const float* W_e2  = (const float*)d_in[4];
  const float* W_a1  = (const float*)d_in[5];
  const float* W_a2  = (const float*)d_in[6];
  const float* W_f1  = (const float*)d_in[7];
  const float* W_f2  = (const float*)d_in[8];
  const float* W_ie1 = (const float*)d_in[9];
  const float* W_ie2 = (const float*)d_in[10];
  const float* W_ie3 = (const float*)d_in[11];
  const float* W_ia1 = (const float*)d_in[12];
  const float* W_ia2 = (const float*)d_in[13];
  const float* W_ia3 = (const float*)d_in[14];
  const float* W_oa1 = (const float*)d_in[15];
  const float* b_oa1 = (const float*)d_in[16];
  const float* W_oa2 = (const float*)d_in[17];
  const float* b_oa2 = (const float*)d_in[18];
  const float* W_at1 = (const float*)d_in[19];
  const float* b_at1 = (const float*)d_in[20];
  const float* W_at2 = (const float*)d_in[21];
  const float* b_at2 = (const float*)d_in[22];
  const float* W_at3 = (const float*)d_in[23];
  const float* b_at3 = (const float*)d_in[24];
  float* out = (float*)d_out;

  // bump allocator over workspace
  char* ws = (char*)d_ws;
  size_t off = 0;
  auto alloc = [&](size_t bytes) -> char* {
    off = (off + 255) & ~(size_t)255;
    char* p = ws + off; off += bytes; return p;
  };
  auto allocH = [&](size_t n) -> _Float16* { return (_Float16*)alloc(n * 2); };
  auto allocF = [&](size_t n) -> float*    { return (float*)alloc(n * 4); };

  // packed weights (transposed, padded) -------------------------------------
  _Float16* We1t  = allocH(64 * 32);
  _Float16* We2t  = allocH(64 * 64);
  _Float16* Wa1t  = allocH(64 * 32);
  _Float16* Wa2t  = allocH(64 * 64);
  _Float16* Wf1t  = allocH(64 * 64);
  _Float16* Wf2t  = allocH(64 * 64);
  _Float16* Wie1t = allocH(64 * 128);
  _Float16* Wie2t = allocH(64 * 64);
  _Float16* Wie3t = allocH(16 * 64);
  _Float16* Wia1t = allocH(64 * 128);
  _Float16* Wia2t = allocH(64 * 64);
  _Float16* Wia3t = allocH(16 * 64);
  _Float16* Woa1t = allocH(64 * 64);
  _Float16* Woa2t = allocH(16 * 64);
  _Float16* Wat1t = allocH(128 * 128);
  _Float16* Wat2t = allocH(64 * 128);
  _Float16* Wat3t = allocH(16 * 64);
  float* bp = allocF(5 * 128);   // padded biases: oa1,oa2,at1,at2,at3
  float* st = allocF(12 * 128);  // qbn stats slots

  // activations --------------------------------------------------------------
  _Float16* ev_h    = allocH((size_t)ME * 32);
  _Float16* av_h    = allocH((size_t)MA * 32);
  _Float16* e_t1_h  = allocH((size_t)ME * 64);
  _Float16* e_rc_h  = allocH((size_t)ME * 64);
  _Float16* a_t1_h  = allocH((size_t)MA * 64);
  _Float16* a_rc_h  = allocH((size_t)MA * 64);
  _Float16* pool_h  = allocH((size_t)MV * 64);
  _Float16* v_t1_h  = allocH((size_t)MV * 64);
  _Float16* vf_h    = allocH((size_t)MV * 64);
  _Float16* sf_h    = allocH((size_t)BNA * 64);
  _Float16* cat_h   = allocH((size_t)ME * 128);
  _Float16* imp1_h  = allocH((size_t)ME * 64);   // shared e/a importance temps
  _Float16* imp2_h  = allocH((size_t)ME * 64);
  _Float16* atf_h   = allocH((size_t)MAT * 128);
  _Float16* ath1_h  = allocH((size_t)MAT * 128);
  _Float16* ath2_h  = allocH((size_t)MAT * 64);
  _Float16* oah1_h  = allocH((size_t)BNA * 64);

  float* Yf    = allocF((size_t)MAT * 128);      // == ME*64 elems, shared GEMM out
  float* e_imp = allocF((size_t)ME);
  float* a_imp = allocF((size_t)ME);
  float* mvb   = allocF((size_t)BNA * 2);
  float* sf_f  = allocF((size_t)BNA * 64);
  float* oaq_f = allocF((size_t)BNA * 16);
  float* aq_f  = allocF((size_t)MAT * 16);

  // --- init stats -----------------------------------------------------------
  zero_f32_k<<<cdiv(12 * 128, 256), 256, 0, stream>>>(st, 12 * 128);

  // --- pack weights & biases ------------------------------------------------
  auto pw = [&](const float* W, _Float16* Wt, int K, int N, int Kp, int Np) {
    pack_w_k<<<cdiv((long)Kp * Np, 256), 256, 0, stream>>>(W, Wt, K, N, Kp, Np);
  };
  pw(W_e1, We1t, 19, 64, 32, 64);   pw(W_e2, We2t, 64, 64, 64, 64);
  pw(W_a1, Wa1t, 19, 64, 32, 64);   pw(W_a2, Wa2t, 64, 64, 64, 64);
  pw(W_f1, Wf1t, 64, 64, 64, 64);   pw(W_f2, Wf2t, 64, 64, 64, 64);
  pw(W_ie1, Wie1t, 128, 64, 128, 64); pw(W_ie2, Wie2t, 64, 64, 64, 64);
  pw(W_ie3, Wie3t, 64, 1, 64, 16);
  pw(W_ia1, Wia1t, 128, 64, 128, 64); pw(W_ia2, Wia2t, 64, 64, 64, 64);
  pw(W_ia3, Wia3t, 64, 1, 64, 16);
  pw(W_oa1, Woa1t, 64, 64, 64, 64); pw(W_oa2, Woa2t, 64, 3, 64, 16);
  pw(W_at1, Wat1t, 128, 128, 128, 128); pw(W_at2, Wat2t, 128, 64, 128, 64);
  pw(W_at3, Wat3t, 64, 1, 64, 16);
  pack_b_k<<<1, 128, 0, stream>>>(b_oa1, bp + 0 * 128, 64, 128);
  pack_b_k<<<1, 128, 0, stream>>>(b_oa2, bp + 1 * 128, 3, 128);
  pack_b_k<<<1, 128, 0, stream>>>(b_at1, bp + 2 * 128, 128, 128);
  pack_b_k<<<1, 128, 0, stream>>>(b_at2, bp + 3 * 128, 64, 128);
  pack_b_k<<<1, 128, 0, stream>>>(b_at3, bp + 4 * 128, 1, 128);

  // --- build_vec ------------------------------------------------------------
  build_vec_k<<<cdiv(BNA * NE, 256), 256, 0, stream>>>(enemy, own, ev_h, NE, BNA * NE);
  build_vec_k<<<cdiv(BNA * NAL, 256), 256, 0, stream>>>(ally, own, av_h, NAL, BNA * NAL);

  auto gemm = [&](const _Float16* X, const _Float16* Wt, float* Y, _Float16* Yh,
                  const float* bias, float* colsq, int M, int K, int N, int relu) {
    dim3 g(M / 16, (N + 63) / 64);
    wmma_gemm<<<g, 128, 0, stream>>>(X, Wt, Y, Yh, bias, colsq, M, K, N, relu);
  };
  auto norm = [&](const float* Y, const float* colsq, int M, int N, int Nvalid,
                  _Float16* oh, float* o32) {
    qbn_qrelu<<<cdiv((long)(M / 2) * Nvalid, 256), 256, 0, stream>>>(Y, colsq, M, N, Nvalid, oh, o32);
  };

  // --- enemy / ally RCN chains ---------------------------------------------
  gemm(ev_h, We1t, Yf, nullptr, nullptr, st + 0 * 128, ME, 32, 64, 0);
  norm(Yf, st + 0 * 128, ME, 64, 64, e_t1_h, nullptr);
  gemm(e_t1_h, We2t, Yf, nullptr, nullptr, st + 1 * 128, ME, 64, 64, 0);
  norm(Yf, st + 1 * 128, ME, 64, 64, e_rc_h, nullptr);

  gemm(av_h, Wa1t, Yf, nullptr, nullptr, st + 2 * 128, MA, 32, 64, 0);
  norm(Yf, st + 2 * 128, MA, 64, 64, a_t1_h, nullptr);
  gemm(a_t1_h, Wa2t, Yf, nullptr, nullptr, st + 3 * 128, MA, 64, 64, 0);
  norm(Yf, st + 3 * 128, MA, 64, 64, a_rc_h, nullptr);

  // --- qmaxpool + vf chain --------------------------------------------------
  qmaxpool_k<<<cdiv(BNA * 64, 256), 256, 0, stream>>>(e_rc_h, a_rc_h, pool_h, BNA * 64);
  gemm(pool_h, Wf1t, Yf, nullptr, nullptr, st + 4 * 128, MV, 64, 64, 0);
  norm(Yf, st + 4 * 128, MV, 64, 64, v_t1_h, nullptr);
  gemm(v_t1_h, Wf2t, Yf, nullptr, nullptr, st + 5 * 128, MV, 64, 64, 0);
  norm(Yf, st + 5 * 128, MV, 64, 64, vf_h, nullptr);

  qmerge_sf_k<<<cdiv(BNA * 64, 256), 256, 0, stream>>>(vf_h, sf_f, sf_h, BNA * 64);
  build_cat_k<<<cdiv((long)ME * 128, 256), 256, 0, stream>>>(vf_h, e_rc_h, cat_h, (long)ME * 128);

  // --- importance chains (e then a, reusing temps) --------------------------
  gemm(cat_h, Wie1t, Yf, nullptr, nullptr, st + 6 * 128, ME, 128, 64, 0);
  norm(Yf, st + 6 * 128, ME, 64, 64, imp1_h, nullptr);
  gemm(imp1_h, Wie2t, Yf, nullptr, nullptr, st + 7 * 128, ME, 64, 64, 0);
  norm(Yf, st + 7 * 128, ME, 64, 64, imp2_h, nullptr);
  gemm(imp2_h, Wie3t, Yf, nullptr, nullptr, st + 8 * 128, ME, 64, 16, 0);
  norm(Yf, st + 8 * 128, ME, 16, 1, nullptr, e_imp);

  gemm(cat_h, Wia1t, Yf, nullptr, nullptr, st + 9 * 128, ME, 128, 64, 0);
  norm(Yf, st + 9 * 128, ME, 64, 64, imp1_h, nullptr);
  gemm(imp1_h, Wia2t, Yf, nullptr, nullptr, st + 10 * 128, ME, 64, 64, 0);
  norm(Yf, st + 10 * 128, ME, 64, 64, imp2_h, nullptr);
  gemm(imp2_h, Wia3t, Yf, nullptr, nullptr, st + 11 * 128, ME, 64, 16, 0);
  norm(Yf, st + 11 * 128, ME, 16, 1, nullptr, a_imp);

  movevec_k<<<cdiv(BNA * 2, 256), 256, 0, stream>>>(e_imp, a_imp, mvb, BNA * 2);

  // --- oaq head (plain ReLU MLP, fused bias+relu in GEMM epilogue) ----------
  gemm(sf_h, Woa1t, nullptr, oah1_h, bp + 0 * 128, nullptr, BNA, 64, 64, 1);
  gemm(oah1_h, Woa2t, oaq_f, nullptr, bp + 1 * 128, nullptr, BNA, 64, 16, 0);

  // --- attention head -------------------------------------------------------
  build_atf_k<<<cdiv((long)MAT * 128, 256), 256, 0, stream>>>(sf_h, e_rc_h, atf_h, (long)MAT * 128);
  gemm(atf_h, Wat1t, nullptr, ath1_h, bp + 2 * 128, nullptr, MAT, 128, 128, 1);
  gemm(ath1_h, Wat2t, nullptr, ath2_h, bp + 3 * 128, nullptr, MAT, 128, 64, 1);
  gemm(ath2_h, Wat3t, aq_f, nullptr, bp + 4 * 128, nullptr, MAT, 64, 16, 0);

  // --- final assemble -------------------------------------------------------
  final_k<<<cdiv(BNA, 256), 256, 0, stream>>>(oaq_f, mvb, aq_f, out, BNA);
}
